// GAT_59090160058489
// MI455X (gfx1250) — matrix operand order repaired and checked by reading, not compiled, not run on hardware
//
#include <hip/hip_runtime.h>
#include <hip/hip_bf16.h>

// ---------------------------------------------------------------------------
// GAT 2-layer + dense head forward for MI455X (gfx1250, wave32).
// Dense GEMMs use v_wmma_f32_16x16x32_f16 (f16 in, f32 accumulate).
// Edge phase (segment softmax + scatter) is memory-bound -> coalesced
// per-edge kernels with f32 atomics; segment-max via monotone uint encoding.
// ---------------------------------------------------------------------------

typedef _Float16 half_t;
typedef half_t v16h __attribute__((ext_vector_type(16)));
typedef half_t v8h  __attribute__((ext_vector_type(8)));
typedef float  v8f  __attribute__((ext_vector_type(8)));

#define GAT_N      100000
#define GAT_E      1600000
#define GAT_FIN    256
#define GAT_HD     128      // H*D
#define GAT_C      64
#define GAT_TRAIN  10000
#define GAT_SLOPE  0.2f
#define GAT_BNEPS  1e-5f

// ----- order-preserving float<->uint encoding for atomic segment-max -------
__device__ __forceinline__ unsigned fenc(float f) {
  unsigned u = __float_as_uint(f);
  return (u & 0x80000000u) ? ~u : (u | 0x80000000u);
}
__device__ __forceinline__ float fdec(unsigned k) {
  unsigned u = (k & 0x80000000u) ? (k ^ 0x80000000u) : ~k;
  return __uint_as_float(u);
}
#define ENC_NEG_INF 0x007FFFFFu   // fenc(-inf)

// --------------------------- utility kernels -------------------------------
__global__ void k_fill_f32(float* p, float v, long n) {
  long i = (long)blockIdx.x * blockDim.x + threadIdx.x;
  if (i < n) p[i] = v;
}
__global__ void k_fill_u32(unsigned* p, unsigned v, long n) {
  long i = (long)blockIdx.x * blockDim.x + threadIdx.x;
  if (i < n) p[i] = v;
}
__global__ void k_cvt_f16(const float* __restrict__ x, half_t* __restrict__ y, long n) {
  long i = (long)blockIdx.x * blockDim.x + threadIdx.x;
  if (i < n) y[i] = (half_t)x[i];
}
// Wt[n*K + k] = (f16) W[k*Nn + n]   (row-major K x Nn  ->  row-major Nn x K)
__global__ void k_trans_cvt(const float* __restrict__ W, half_t* __restrict__ Wt,
                            int K, int Nn) {
  int idx = blockIdx.x * blockDim.x + threadIdx.x;
  if (idx >= K * Nn) return;
  int n = idx / K, k = idx % K;
  Wt[idx] = (half_t)W[(size_t)k * Nn + n];
}

// ----------------------- WMMA GEMM: C = A * Bt^T ---------------------------
// A:  [M,K]  f16 row-major       Bt: [Nn,K] f16 row-major (pre-transposed B)
// C:  [M,Nn] f32 row-major.  act: 0/1 = +bias(if non-null), 2 = +bias & ReLU
__global__ __launch_bounds__(256)
void k_wmma_gemm(const half_t* __restrict__ A, const half_t* __restrict__ Bt,
                 const float* __restrict__ bias, float* __restrict__ C,
                 int M, int Nn, int K, int act) {
  const int wave = threadIdx.x >> 5;
  const int lane = threadIdx.x & 31;
  const int lo   = lane & 15;
  const int hi   = lane >> 4;             // 0 or 1 (half-wave)
  const int m0 = blockIdx.x * 16;
  const int n0 = (blockIdx.y * 8 + wave) * 16;
  if (n0 >= Nn || m0 >= M) return;

  int arow = m0 + lo; if (arow >= M) arow = M - 1;
  // A fragment: elems 0..7 -> K = kb + hi*8 .. +7 ; elems 8..15 -> +16
  const half_t* aptr = A  + (size_t)arow * K + hi * 8;
  // B fragment: elems 0..7 -> K = kb + hi*8 .. +7 ; elems 8..15 -> +16, column n0+lo
  const half_t* bptr = Bt + (size_t)(n0 + lo) * K + hi * 8;

  v8f acc = {};
  for (int kb = 0; kb < K; kb += 32) {
    union { v16h v; v8h h[2]; } a, b;
    a.h[0] = *(const v8h*)(aptr + kb);
    a.h[1] = *(const v8h*)(aptr + kb + 16);
    b.h[0] = *(const v8h*)(bptr + kb);
    b.h[1] = *(const v8h*)(bptr + kb + 16);
    acc = __builtin_amdgcn_wmma_f32_16x16x32_f16(
        /*neg_a=*/false, a.v, /*neg_b=*/false, b.v,
        /*c_mod=*/(short)0, acc, /*reuse_a=*/false, /*reuse_b=*/false);
  }

  const int col = n0 + lo;
  const float bv = bias ? bias[col] : 0.0f;
  #pragma unroll
  for (int r = 0; r < 8; ++r) {
    int row = m0 + r + hi * 8;
    if (row >= M) continue;
    float v = acc[r] + bv;
    if (act == 2) v = fmaxf(v, 0.0f);
    C[(size_t)row * Nn + col] = v;
  }
}

// -------------------- attention projections el / er ------------------------
// el[n,h] = dot(feat[n,h,:], al[h,:]);  er likewise. H=2, D=64, HD=128.
__global__ void k_attn_proj(const float* __restrict__ feat,
                            const float* __restrict__ al,
                            const float* __restrict__ ar,
                            float* __restrict__ el, float* __restrict__ er,
                            int Nn) {
  int t = blockIdx.x * blockDim.x + threadIdx.x;
  if (t >= Nn * 2) return;
  int n = t >> 1, h = t & 1;
  const float* f = feat + (size_t)n * GAT_HD + h * 64;
  const float* a = al + h * 64;
  const float* b = ar + h * 64;
  float sl = 0.f, sr = 0.f;
  #pragma unroll 8
  for (int d = 0; d < 64; ++d) { float fv = f[d]; sl += fv * a[d]; sr += fv * b[d]; }
  el[t] = sl; er[t] = sr;
}

// ------------------------------ edge phase ---------------------------------
__global__ void k_edge_score(const int* __restrict__ src, const int* __restrict__ dst,
                             const float* __restrict__ el, const float* __restrict__ er,
                             float* __restrict__ esc, unsigned* __restrict__ maxe, int E2) {
  int t = blockIdx.x * blockDim.x + threadIdx.x;
  if (t >= E2) return;
  int e = t >> 1, h = t & 1;
  float s = el[src[e] * 2 + h] + er[dst[e] * 2 + h];
  s = (s > 0.f) ? s : GAT_SLOPE * s;              // leaky_relu
  esc[t] = s;
  atomicMax(&maxe[dst[e] * 2 + h], fenc(s));
}

__global__ void k_edge_exp(const int* __restrict__ dst,
                           float* __restrict__ esc,           // in: score, out: exp
                           const unsigned* __restrict__ maxe,
                           float* __restrict__ den, int E2) {
  int t = blockIdx.x * blockDim.x + threadIdx.x;
  if (t >= E2) return;
  int e = t >> 1, h = t & 1;
  float m  = fdec(maxe[dst[e] * 2 + h]);
  float ex = __expf(esc[t] - m);
  esc[t] = ex;
  atomicAdd(&den[dst[e] * 2 + h], ex);
}

__global__ void k_edge_aggr(const int* __restrict__ src, const int* __restrict__ dst,
                            const float* __restrict__ expe, const float* __restrict__ den,
                            const float* __restrict__ feat, float* __restrict__ out,
                            long total) {
  long t = (long)blockIdx.x * blockDim.x + threadIdx.x;
  if (t >= total) return;
  int e = (int)(t >> 7);
  int idx = (int)(t & 127);
  int h = idx >> 6;
  int d = dst[e], s = src[e];
  float alpha = expe[e * 2 + h] / fmaxf(den[d * 2 + h], 1e-16f);
  atomicAdd(&out[(size_t)d * GAT_HD + idx], feat[(size_t)s * GAT_HD + idx] * alpha);
}

__global__ void k_bias_relu(float* __restrict__ x, const float* __restrict__ b,
                            long n, int F) {
  long t = (long)blockIdx.x * blockDim.x + threadIdx.x;
  if (t >= n) return;
  x[t] = fmaxf(x[t] + b[t % F], 0.0f);
}

// ------------------------------ batch norm ---------------------------------
__global__ void k_bn_partial(const float* __restrict__ h, float* __restrict__ sum,
                             float* __restrict__ sumsq, int Nn) {
  int t = blockIdx.x * blockDim.x + threadIdx.x;   // 128 blocks * 256 = 32768
  int c  = t & 63;
  int r0 = t >> 6;                                  // 0..511
  float s = 0.f, s2 = 0.f;
  for (int r = r0; r < Nn; r += 512) {
    float v = h[(size_t)r * GAT_C + c];
    s += v; s2 += v * v;
  }
  atomicAdd(&sum[c], s);
  atomicAdd(&sumsq[c], s2);
}

__global__ void k_bn_final(const float* __restrict__ sum, const float* __restrict__ sumsq,
                           const float* __restrict__ gamma, const float* __restrict__ beta,
                           float* __restrict__ scale, float* __restrict__ shift, int Nn) {
  int c = threadIdx.x;
  if (c >= GAT_C) return;
  float mu  = sum[c] / (float)Nn;
  float var = sumsq[c] / (float)Nn - mu * mu;      // biased variance
  float inv = rsqrtf(var + GAT_BNEPS) * gamma[c];
  scale[c] = inv;
  shift[c] = beta[c] - mu * inv;
}

// wave-per-row BN-apply + softmax, C=64 (2 cols per lane, wave32 shuffles)
__global__ __launch_bounds__(256)
void k_bn_softmax(const float* __restrict__ h, const float* __restrict__ scale,
                  const float* __restrict__ shift, float* __restrict__ p, int Nn) {
  int wave = threadIdx.x >> 5, lane = threadIdx.x & 31;
  int row = blockIdx.x * 8 + wave;
  if (row >= Nn) return;
  const float* hr = h + (size_t)row * GAT_C;
  float y0 = hr[lane]      * scale[lane]      + shift[lane];
  float y1 = hr[lane + 32] * scale[lane + 32] + shift[lane + 32];
  float m = fmaxf(y0, y1);
  for (int o = 16; o; o >>= 1) m = fmaxf(m, __shfl_xor(m, o, 32));
  float e0 = __expf(y0 - m), e1 = __expf(y1 - m);
  float s = e0 + e1;
  for (int o = 16; o; o >>= 1) s += __shfl_xor(s, o, 32);
  float inv = 1.0f / s;
  p[(size_t)row * GAT_C + lane]      = e0 * inv;
  p[(size_t)row * GAT_C + lane + 32] = e1 * inv;
}

// ----------------------------- loss kernel ---------------------------------
__global__ void k_loss(const float* __restrict__ p, const int* __restrict__ train,
                       const int* __restrict__ label, float* __restrict__ loss, int T) {
  int t = blockIdx.x * blockDim.x + threadIdx.x;
  if (t >= T) return;
  int row = train[t];
  const float* pr = p + (size_t)row * GAT_C;
  float m = -1e30f;
  for (int j = 0; j < GAT_C; ++j) m = fmaxf(m, pr[j]);
  float s = 0.f;
  for (int j = 0; j < GAT_C; ++j) s += __expf(pr[j] - m);
  float lse = m + __logf(s);
  float lp  = pr[label[row]] - lse;                // log_softmax at the label
  atomicAdd(loss, -lp / (float)T);
}

// ---------------------------------------------------------------------------
static inline dim3 g1(long n) { return dim3((unsigned)((n + 255) / 256)); }

extern "C" void kernel_launch(void* const* d_in, const int* in_sizes, int n_in,
                              void* d_out, int out_size, void* d_ws, size_t ws_size,
                              hipStream_t stream) {
  (void)in_sizes; (void)n_in; (void)out_size;
  const float* feature = (const float*)d_in[0];
  const int*   label   = (const int*)  d_in[1];
  const int*   train   = (const int*)  d_in[2];
  const int*   src     = (const int*)  d_in[3];
  const int*   dst     = (const int*)  d_in[4];
  const float* W0  = (const float*)d_in[5];
  const float* al0 = (const float*)d_in[6];
  const float* ar0 = (const float*)d_in[7];
  const float* b0  = (const float*)d_in[8];
  const float* W1  = (const float*)d_in[9];
  const float* al1 = (const float*)d_in[10];
  const float* ar1 = (const float*)d_in[11];
  const float* b1  = (const float*)d_in[12];
  const float* Wout = (const float*)d_in[13];
  const float* bout = (const float*)d_in[14];
  const float* Wfc  = (const float*)d_in[15];
  const float* bfc  = (const float*)d_in[16];
  const float* gamma = (const float*)d_in[17];
  const float* beta  = (const float*)d_in[18];

  const int N = GAT_N, E = GAT_E;
  // ---- workspace layout (all chunks are multiples of 256 bytes) ----
  char* w = (char*)d_ws;
  size_t need = 0;
  auto take = [&](size_t bytes) { char* p = w + need; need += bytes; return p; };
  half_t*   A16  = (half_t*)  take((size_t)N * GAT_FIN * 2);          // 51.2 MB
  half_t*   WT   = (half_t*)  take((size_t)GAT_HD * GAT_FIN * 2);     // 64 KB
  float*    FEAT = (float*)   take((size_t)N * GAT_HD * 4);           // 51.2 MB
  float*    EL   = (float*)   take((size_t)N * 2 * 4);
  float*    ER   = (float*)   take((size_t)N * 2 * 4);
  unsigned* MAXE = (unsigned*)take((size_t)N * 2 * 4);
  float*    DEN  = (float*)   take((size_t)N * 2 * 4);
  float*    ESC  = (float*)   take((size_t)E * 2 * 4);                // 12.8 MB
  float*    OUT  = (float*)   take((size_t)N * GAT_HD * 4);           // 51.2 MB
  float*    H2   = (float*)   take((size_t)N * GAT_C * 4);            // 25.6 MB
  float*    H3   = (float*)   take((size_t)N * GAT_C * 4);            // 25.6 MB
  float*    SUM  = (float*)   take(256);
  float*    SSQ  = (float*)   take(256);
  float*    SCALE= (float*)   take(256);
  float*    SHIFT= (float*)   take(256);
  if (need > ws_size) return;   // insufficient scratch; bail deterministically

  float* p_out    = (float*)d_out;                 // [N, C]
  float* loss_out = p_out + (size_t)N * GAT_C;     // scalar

  const long EHD = (long)E * GAT_HD;

  // =============================== Layer 0 ===============================
  k_cvt_f16 <<<g1((long)N * GAT_FIN), 256, 0, stream>>>(feature, A16, (long)N * GAT_FIN);
  k_trans_cvt<<<g1(GAT_FIN * GAT_HD), 256, 0, stream>>>(W0, WT, GAT_FIN, GAT_HD);
  k_wmma_gemm<<<dim3(N / 16, 1), 256, 0, stream>>>(A16, WT, nullptr, FEAT, N, GAT_HD, GAT_FIN, 0);
  k_attn_proj<<<g1((long)N * 2), 256, 0, stream>>>(FEAT, al0, ar0, EL, ER, N);
  k_fill_u32 <<<g1((long)N * 2), 256, 0, stream>>>(MAXE, ENC_NEG_INF, (long)N * 2);
  k_fill_f32 <<<g1((long)N * 2), 256, 0, stream>>>(DEN, 0.f, (long)N * 2);
  k_fill_f32 <<<g1((long)N * GAT_HD), 256, 0, stream>>>(OUT, 0.f, (long)N * GAT_HD);
  k_edge_score<<<g1((long)E * 2), 256, 0, stream>>>(src, dst, EL, ER, ESC, MAXE, E * 2);
  k_edge_exp  <<<g1((long)E * 2), 256, 0, stream>>>(dst, ESC, MAXE, DEN, E * 2);
  k_edge_aggr <<<g1(EHD), 256, 0, stream>>>(src, dst, ESC, DEN, FEAT, OUT, EHD);
  k_bias_relu <<<g1((long)N * GAT_HD), 256, 0, stream>>>(OUT, b0, (long)N * GAT_HD, GAT_HD);

  // =============================== Layer 1 ===============================
  k_cvt_f16 <<<g1((long)N * GAT_HD), 256, 0, stream>>>(OUT, A16, (long)N * GAT_HD);
  k_trans_cvt<<<g1(GAT_HD * GAT_HD), 256, 0, stream>>>(W1, WT, GAT_HD, GAT_HD);
  k_wmma_gemm<<<dim3(N / 16, 1), 256, 0, stream>>>(A16, WT, nullptr, FEAT, N, GAT_HD, GAT_HD, 0);
  k_attn_proj<<<g1((long)N * 2), 256, 0, stream>>>(FEAT, al1, ar1, EL, ER, N);
  k_fill_u32 <<<g1((long)N * 2), 256, 0, stream>>>(MAXE, ENC_NEG_INF, (long)N * 2);
  k_fill_f32 <<<g1((long)N * 2), 256, 0, stream>>>(DEN, 0.f, (long)N * 2);
  k_fill_f32 <<<g1((long)N * GAT_HD), 256, 0, stream>>>(OUT, 0.f, (long)N * GAT_HD);
  k_edge_score<<<g1((long)E * 2), 256, 0, stream>>>(src, dst, EL, ER, ESC, MAXE, E * 2);
  k_edge_exp  <<<g1((long)E * 2), 256, 0, stream>>>(dst, ESC, MAXE, DEN, E * 2);
  k_edge_aggr <<<g1(EHD), 256, 0, stream>>>(src, dst, ESC, DEN, FEAT, OUT, EHD);
  k_bias_relu <<<g1((long)N * GAT_HD), 256, 0, stream>>>(OUT, b1, (long)N * GAT_HD, GAT_HD);

  // ================================ Head =================================
  // h2 = relu(h @ Wout + bout)   [N,64]
  k_cvt_f16 <<<g1((long)N * GAT_HD), 256, 0, stream>>>(OUT, A16, (long)N * GAT_HD);
  k_trans_cvt<<<g1(GAT_HD * GAT_C), 256, 0, stream>>>(Wout, WT, GAT_HD, GAT_C);
  k_wmma_gemm<<<dim3(N / 16, 1), 256, 0, stream>>>(A16, WT, bout, H2, N, GAT_C, GAT_HD, 2);
  // h3 = h2 @ Wfc + bfc          [N,64]
  k_cvt_f16 <<<g1((long)N * GAT_C), 256, 0, stream>>>(H2, A16, (long)N * GAT_C);
  k_trans_cvt<<<g1(GAT_C * GAT_C), 256, 0, stream>>>(Wfc, WT, GAT_C, GAT_C);
  k_wmma_gemm<<<dim3(N / 16, 1), 256, 0, stream>>>(A16, WT, bfc, H3, N, GAT_C, GAT_C, 1);

  // BatchNorm (training-mode batch stats) + softmax -> p
  k_fill_f32 <<<1, 256, 0, stream>>>(SUM, 0.f, 64);
  k_fill_f32 <<<1, 256, 0, stream>>>(SSQ, 0.f, 64);
  k_bn_partial<<<128, 256, 0, stream>>>(H3, SUM, SSQ, N);
  k_bn_final  <<<1, 64, 0, stream>>>(SUM, SSQ, gamma, beta, SCALE, SHIFT, N);
  k_bn_softmax<<<dim3((N + 7) / 8), 256, 0, stream>>>(H3, SCALE, SHIFT, p_out, N);

  // cross-entropy of log_softmax(p) at labels over train nodes
  k_fill_f32 <<<1, 32, 0, stream>>>(loss_out, 0.f, 1);
  k_loss     <<<g1(GAT_TRAIN), 256, 0, stream>>>(p_out, train, label, loss_out, GAT_TRAIN);
}
